// VoxelResBackBone8xVoxelNeXt2DSPS_58428735095120
// MI455X (gfx1250) — compile-verified, matrix operand order checked by reading, not run on hardware
//
#include <hip/hip_runtime.h>

// ---------------------------------------------------------------------------
// Types for CDNA5 WMMA (gfx1250, wave32)
// ---------------------------------------------------------------------------
typedef __bf16 bf16_t;
typedef __attribute__((ext_vector_type(16))) __bf16 v16bf;
typedef __attribute__((ext_vector_type(8)))  float  v8f;

struct Frag32B { uint4 lo, hi; };   // 32 bytes == one v16bf fragment

#define HISTBINS 4096

__device__ __forceinline__ unsigned short f2bfbits(float f) {
    unsigned u = __builtin_bit_cast(unsigned, f);
    unsigned r = u + 0x7fffu + ((u >> 16) & 1u);   // round-to-nearest-even
    return (unsigned short)(r >> 16);
}
__device__ __forceinline__ bf16_t f2bf(float f) {
    unsigned short h = f2bfbits(f);
    return __builtin_bit_cast(bf16_t, h);
}
__device__ __forceinline__ float bf2f(bf16_t h) {
    unsigned short u = __builtin_bit_cast(unsigned short, h);
    unsigned x = ((unsigned)u) << 16;
    return __builtin_bit_cast(float, x);
}

// ---------------------------------------------------------------------------
// Weight pre-pack: fp32 (O,I,3,3) -> bf16 B-fragments, layout
//   [tap][ktile][ntile][lane][16]   (lane/elem mapping per ISA 16-bit layout)
// For lane L, elem e: v=e/2, el=e&1
//   K = (v<4?0:16) + (L>>4)*8 + (v&3)*2 + el + 32*ktile
//   N = (L&15) + 16*ntile
// ---------------------------------------------------------------------------
__global__ void pack_weights_k(const float* __restrict__ w,
                               unsigned short* __restrict__ out,
                               int Cin, int Cout) {
    int Kt = Cin >> 5, Nt = Cout >> 4;
    int total = 9 * Kt * Nt * 512;
    int i = blockIdx.x * 256 + threadIdx.x;
    if (i >= total) return;
    int e    = i & 15;
    int lane = (i >> 4) & 31;
    int frag = i >> 9;
    int nt = frag % Nt; frag /= Nt;
    int kt = frag % Kt;
    int tap = frag / Kt;
    int v = e >> 1, el = e & 1;
    int K = ((v < 4) ? 0 : 16) + ((lane >> 4) << 3) + ((v & 3) << 1) + el + (kt << 5);
    int N = (nt << 4) + (lane & 15);
    float f = w[((long)N * Cin + K) * 9 + tap];
    out[i] = f2bfbits(f);
}

// fp32 NCHW -> bf16 NHWC (pixel-major, channel-minor) ingest transform.
__global__ void nchw2nhwc_bf16_k(const float* __restrict__ in,
                                 bf16_t* __restrict__ out, int C, long HW) {
    long i = (long)blockIdx.x * 256 + threadIdx.x;
    if (i >= HW * C) return;
    int c = (int)(i % C);
    long p = i / C;
    out[i] = f2bf(in[(long)c * HW + p]);
}

// out[p,c] = keep[p] ? in[p,c] : 0   (bf16 NHWC; builds the x*kept view)
__global__ void maskmul_k(const bf16_t* __restrict__ in,
                          const float* __restrict__ keep,
                          bf16_t* __restrict__ out, int C, long total) {
    long i = (long)blockIdx.x * 256 + threadIdx.x;
    if (i >= total) return;
    long p = i / C;
    unsigned short z = 0;
    out[i] = (keep[p] > 0.0f) ? in[i] : __builtin_bit_cast(bf16_t, z);
}

// ---------------------------------------------------------------------------
// Fused implicit-GEMM 3x3 conv (pad=1), activations bf16 NHWC.
// Templated on KT = Cin/32 so ALL loops fully unroll: the unroller acts as
// the software pipeliner (register renaming instead of mov-rotation; the
// scheduler hoists next-iteration global_load_b128 / ds_load_b128 above the
// current v_wmma with partial waits). Per-tap bases + inbounds flags are
// precomputed; tap constants fold (incl. the SPS center-tap pointer select).
// Block = 128 threads = 4 waves x 1 M-tile; each wave computes TWO N-tiles
// (32 out channels) per A fragment. B staged in LDS 3 taps at a time
// (<= 48KB). A loads are unconditional (row-0-safe) and zeroed by v_cndmask
// at consumption.
// ---------------------------------------------------------------------------
template <int KT>
__global__ __launch_bounds__(128)
void conv3x3_wmma(const bf16_t* __restrict__ inCenter,
                  const bf16_t* __restrict__ inOther,
                  const unsigned short* __restrict__ wpk,
                  const float* __restrict__ bias,
                  const float* __restrict__ bng, const float* __restrict__ bnb,
                  const float* __restrict__ bnm, const float* __restrict__ bnv,
                  float eps,
                  const bf16_t* __restrict__ residual,
                  const float* __restrict__ maskOut,
                  bf16_t* __restrict__ outBF, float* __restrict__ outF32,
                  int Cout, int Hin, int Win,
                  int Hout, int Wout, int stride) {
    extern __shared__ unsigned short ldsB[];
    constexpr int Cin = KT * 32;
    const int Nt = Cout >> 4;
    const int nt0 = blockIdx.y * 2;
    const int tid = threadIdx.x;
    const int lane = tid & 31;
    const int wave = tid >> 5;
    const int tileM = blockIdx.x * 4 + wave;
    const int HWo = Hout * Wout;
    const int half = lane >> 4;
    const int pix = tileM * 16 + (lane & 15);
    const bool valid = pix < HWo;
    const int oh = valid ? pix / Wout : 0;
    const int ow = valid ? pix % Wout : 0;

    // Per-tap base addresses (row-0-safe when OOB) + inbounds flags.
    const bf16_t* base[9];
    bool binb[9];
#pragma unroll
    for (int t = 0; t < 9; ++t) {
        const int ty = t / 3, tx = t - ty * 3;
        const int ih = oh * stride + ty - 1;
        const int iw = ow * stride + tx - 1;
        const bool ib = valid && ih >= 0 && ih < Hin && iw >= 0 && iw < Win;
        const bf16_t* inp = (t == 4) ? inCenter : inOther;   // constant-folds
        long ro = ib ? ((long)ih * Win + iw) * (long)Cin : 0;
        base[t] = inp + ro;
        binb[t] = ib;
    }

    v8f acc0 = {}, acc1 = {};

#pragma unroll
    for (int tg = 0; tg < 3; ++tg) {
        // ---- stage B fragments for taps [3tg, 3tg+3) x {nt0, nt0+1} ------
        __syncthreads();     // previous group's LDS reads complete
        {
            constexpr int nvec = 3 * KT * 2 * 64;    // uint4s (1KB per chunk/nt)
            const int chunk0 = tg * 3 * KT;          // global chunk = tap*KT+kt
            uint4* dst = (uint4*)ldsB;
#pragma unroll
            for (int t0 = 0; t0 < nvec; t0 += 128) {
                int t = t0 + tid;
                int c   = t >> 7;
                int ntl = (t >> 6) & 1;
                int r   = t & 63;
                const uint4* g = (const uint4*)
                    (wpk + (((long)(chunk0 + c) * Nt + nt0 + ntl) << 9)) + r;
                dst[t] = *g;
            }
        }
        __syncthreads();

#pragma unroll
        for (int tl = 0; tl < 3; ++tl) {
            const int tap = tg * 3 + tl;             // constant after unroll
#pragma unroll
            for (int kt = 0; kt < KT; ++kt) {
                const int lc = tl * KT + kt;
                // A fragment: two unconditional b128 loads, cndmask-zeroed
                const int c0 = (kt << 5) + (half << 3);
                const uint4 lo = *(const uint4*)(base[tap] + c0);
                const uint4 hi = *(const uint4*)(base[tap] + c0 + 16);
                const bool ib = binb[tap];
                Frag32B fa;
                fa.lo.x = ib ? lo.x : 0u;  fa.lo.y = ib ? lo.y : 0u;
                fa.lo.z = ib ? lo.z : 0u;  fa.lo.w = ib ? lo.w : 0u;
                fa.hi.x = ib ? hi.x : 0u;  fa.hi.y = ib ? hi.y : 0u;
                fa.hi.z = ib ? hi.z : 0u;  fa.hi.w = ib ? hi.w : 0u;
                v16bf a = __builtin_bit_cast(v16bf, fa);

                // B fragments from LDS (ds_load_b128 x4)
                const uint4* b0p = (const uint4*)(ldsB + (((lc * 2 + 0) * 32 + lane) << 4));
                const uint4* b1p = (const uint4*)(ldsB + (((lc * 2 + 1) * 32 + lane) << 4));
                Frag32B fb0; fb0.lo = b0p[0]; fb0.hi = b0p[1];
                Frag32B fb1; fb1.lo = b1p[0]; fb1.hi = b1p[1];
                v16bf b0 = __builtin_bit_cast(v16bf, fb0);
                v16bf b1 = __builtin_bit_cast(v16bf, fb1);

                acc0 = __builtin_amdgcn_wmma_f32_16x16x32_bf16(
                    false, a, false, b0, (short)0, acc0, false, false);
                acc1 = __builtin_amdgcn_wmma_f32_16x16x32_bf16(
                    false, a, false, b1, (short)0, acc1, false, false);
            }
        }
    }

    // ---- epilogue: C/D layout -> (M = v + 8*half, N = lane&15), 2 N-tiles -
    const int nl = lane & 15;
    const int n0 = (nt0 << 4) + nl;
    const int n1 = n0 + 16;
    const float sc0 = bng[n0] * rsqrtf(bnv[n0] + eps);
    const float sh0 = bnb[n0] - bnm[n0] * sc0;
    const float sc1 = bng[n1] * rsqrtf(bnv[n1] + eps);
    const float sh1 = bnb[n1] - bnm[n1] * sc1;
    const float bs0 = bias ? bias[n0] : 0.0f;
    const float bs1 = bias ? bias[n1] : 0.0f;
#pragma unroll
    for (int v = 0; v < 8; ++v) {
        int p = (tileM << 4) + v + (half << 3);
        if (p >= HWo) continue;
        float v0 = (acc0[v] + bs0) * sc0 + sh0;
        float v1 = (acc1[v] + bs1) * sc1 + sh1;
        if (residual) {
            v0 += bf2f(residual[(long)p * Cout + n0]);
            v1 += bf2f(residual[(long)p * Cout + n1]);
        }
        v0 = fmaxf(v0, 0.0f);
        v1 = fmaxf(v1, 0.0f);
        if (maskOut) { float mm = maskOut[p]; v0 *= mm; v1 *= mm; }
        if (outF32) {                       // fp32 NCHW (final network output)
            outF32[(long)n0 * HWo + p] = v0;
            outF32[(long)n1 * HWo + p] = v1;
        } else {                            // bf16 NHWC (next layer's input)
            outBF[(long)p * Cout + n0] = f2bf(v0);
            outBF[(long)p * Cout + n1] = f2bf(v1);
        }
    }
}

// ---------------------------------------------------------------------------
// Active-set dilation of a 3x3 (s=1 or 2, pad 1) sparse conv (f32 masks).
// ---------------------------------------------------------------------------
__global__ void dilate_k(const float* __restrict__ m, float* __restrict__ out,
                         int Hin, int Hout, int stride) {
    int i = blockIdx.x * 256 + threadIdx.x;
    if (i >= Hout * Hout) return;
    int oh = i / Hout, ow = i % Hout;
    float r = 0.0f;
    for (int ty = 0; ty < 3; ++ty)
        for (int tx = 0; tx < 3; ++tx) {
            int ih = oh * stride + ty - 1, iw = ow * stride + tx - 1;
            if (ih >= 0 && ih < Hin && iw >= 0 && iw < Hin &&
                m[ih * Hin + iw] > 0.0f) r = 1.0f;
        }
    out[i] = r;
}

// Per-voxel sigmoid importance (bf16 NHWC input, contiguous channel dot).
__global__ void score_k(const bf16_t* __restrict__ x, const float* __restrict__ mask,
                        const float* __restrict__ wat, const float* __restrict__ bat,
                        float* __restrict__ score, int* __restrict__ hist,
                        float* __restrict__ cnt, int C, int HW) {
    int i = blockIdx.x * 256 + threadIdx.x;
    if (i >= HW) return;
    float s = 0.0f;
    for (int c = 0; c < C; ++c) s += bf2f(x[(long)i * C + c]) * wat[c];
    s = 1.0f / (1.0f + expf(-(s + bat[0])));
    score[i] = s;
    if (mask[i] > 0.0f) {
        atomicAdd(cnt, 1.0f);
        int b = (int)(s * HISTBINS);
        b = b < 0 ? 0 : (b > HISTBINS - 1 ? HISTBINS - 1 : b);
        atomicAdd(hist + b, 1);
    }
}

// Descending scan of histogram -> threshold for "keep top (1-ratio)*n".
__global__ void thresh_k(const int* __restrict__ hist, const float* __restrict__ cnt,
                         float ratio, float* __restrict__ thr) {
    if (threadIdx.x || blockIdx.x) return;
    int k = (int)(cnt[0] * (1.0f - ratio));
    if (k < 1) k = 1;
    int acc = 0, b = HISTBINS - 1;
    for (; b > 0; --b) { acc += hist[b]; if (acc >= k) break; }
    thr[0] = (float)b / (float)HISTBINS;
}

__global__ void kept_k(const float* __restrict__ mask, const float* __restrict__ score,
                       const float* __restrict__ thr, float* __restrict__ kept,
                       float* __restrict__ unkept, int HW) {
    int i = blockIdx.x * 256 + threadIdx.x;
    if (i >= HW) return;
    float m = mask[i];
    float kk = (m > 0.0f && score[i] >= thr[0]) ? 1.0f : 0.0f;
    kept[i] = kk;
    unkept[i] = (m > 0.0f && kk == 0.0f) ? 1.0f : 0.0f;
}

// m_out = clip(dilate(kept,2,1) + unkept[::2,::2], 0, 1)
__global__ void dynmask_k(const float* __restrict__ kept, const float* __restrict__ unkept,
                          float* __restrict__ mout, int Hin, int Hout) {
    int i = blockIdx.x * 256 + threadIdx.x;
    if (i >= Hout * Hout) return;
    int oh = i / Hout, ow = i % Hout;
    float r = 0.0f;
    for (int ty = 0; ty < 3; ++ty)
        for (int tx = 0; tx < 3; ++tx) {
            int ih = 2 * oh + ty - 1, iw = 2 * ow + tx - 1;
            if (ih >= 0 && ih < Hin && iw >= 0 && iw < Hin &&
                kept[ih * Hin + iw] > 0.0f) r = 1.0f;
        }
    if (unkept[(2 * oh) * Hin + 2 * ow] > 0.0f) r = 1.0f;
    mout[i] = r;
}

// BEV scatter-add of stride-16/32 features into the stride-8 grid (bf16 NHWC).
__global__ void bev_add_k(const bf16_t* __restrict__ x4, const bf16_t* __restrict__ x5,
                          const bf16_t* __restrict__ x6, bf16_t* __restrict__ xb,
                          int C, int H) {
    long i = (long)blockIdx.x * 256 + threadIdx.x;
    long total = (long)H * H * C;
    if (i >= total) return;
    int c = (int)(i % C);
    long p = i / C;
    int oh = (int)(p / H), ow = (int)(p % H);
    int H2 = H >> 1, H4 = H >> 2;
    float v = bf2f(x4[i]);
    if (((oh | ow) & 1) == 0)
        v += bf2f(x5[((long)(oh >> 1) * H2 + (ow >> 1)) * C + c]);
    if (((oh | ow) & 3) == 0)
        v += bf2f(x6[((long)(oh >> 2) * H4 + (ow >> 2)) * C + c]);
    xb[i] = f2bf(v);
}

__global__ void bev_mask_k(const float* __restrict__ m4, const float* __restrict__ m5,
                           const float* __restrict__ m6, float* __restrict__ mb, int H) {
    int i = blockIdx.x * 256 + threadIdx.x;
    if (i >= H * H) return;
    int oh = i / H, ow = i % H;
    int H2 = H >> 1, H4 = H >> 2;
    float v = m4[i];
    if (((oh | ow) & 1) == 0) v = fmaxf(v, m5[(oh >> 1) * H2 + (ow >> 1)]);
    if (((oh | ow) & 3) == 0) v = fmaxf(v, m6[(oh >> 2) * H4 + (ow >> 2)]);
    mb[i] = v;
}

// ---------------------------------------------------------------------------
// Host orchestration
// ---------------------------------------------------------------------------
extern "C" void kernel_launch(void* const* d_in, const int* in_sizes, int n_in,
                              void* d_out, int out_size, void* d_ws, size_t ws_size,
                              hipStream_t stream) {
    (void)in_sizes; (void)n_in; (void)out_size; (void)ws_size;

    const float* x0f   = (const float*)d_in[0];   // (1,32,736,736) fp32 NCHW
    const float* mask1 = (const float*)d_in[1];   // (1,1,736,736)

    int idx = 2;
    auto next = [&]() -> const float* { return (const float*)d_in[idx++]; };

    struct BNP { const float *g, *b, *m, *v; };
    struct BBP { const float *w1, *b1; BNP bn1; const float *w2, *b2; BNP bn2;
                 unsigned short *p1, *p2; };
    struct DWP { const float* w; BNP bn; const float *wat, *bat;
                 unsigned short* p; };

    auto rBN = [&]() -> BNP { BNP r; r.g = next(); r.b = next(); r.m = next(); r.v = next(); return r; };
    auto rBB = [&]() -> BBP { BBP r{}; r.w1 = next(); r.b1 = next(); r.bn1 = rBN();
                              r.w2 = next(); r.b2 = next(); r.bn2 = rBN(); return r; };
    auto rDW = [&](bool attn) -> DWP { DWP r{}; r.w = next(); r.bn = rBN();
                                       if (attn) { r.wat = next(); r.bat = next(); } return r; };

    BBP c1[2]; c1[0] = rBB(); c1[1] = rBB();
    DWP d2 = rDW(true);  BBP c2[2]; c2[0] = rBB(); c2[1] = rBB();
    DWP d3 = rDW(true);  BBP c3[2]; c3[0] = rBB(); c3[1] = rBB();
    DWP d4 = rDW(true);  BBP c4[2]; c4[0] = rBB(); c4[1] = rBB();
    DWP d5 = rDW(false); BBP c5[2]; c5[0] = rBB(); c5[1] = rBB();
    DWP d6 = rDW(false); BBP c6[2]; c6[0] = rBB(); c6[1] = rBB();
    const float* wout_w = next(); BNP wout_bn = rBN();
    const float* sh_w = next(); const float* sh_b = next(); BNP sh_bn = rBN();

    // ---- bump allocator over workspace -----------------------------------
    char* base = (char*)d_ws;
    size_t off = 0;
    auto alloc = [&](size_t nb) -> void* {
        void* p = base + off; off += (nb + 255) & ~(size_t)255; return p;
    };

    auto packw = [&](const float* w, int Cin, int Cout) -> unsigned short* {
        unsigned short* p = (unsigned short*)alloc((size_t)9 * Cin * Cout * 2);
        int tot = 9 * Cin * Cout;
        pack_weights_k<<<dim3((tot + 255) / 256), dim3(256), 0, stream>>>(w, p, Cin, Cout);
        return p;
    };

    for (int i = 0; i < 2; ++i) { c1[i].p1 = packw(c1[i].w1, 32, 32);  c1[i].p2 = packw(c1[i].w2, 32, 32); }
    d2.p = packw(d2.w, 32, 64);
    for (int i = 0; i < 2; ++i) { c2[i].p1 = packw(c2[i].w1, 64, 64);  c2[i].p2 = packw(c2[i].w2, 64, 64); }
    d3.p = packw(d3.w, 64, 128);
    for (int i = 0; i < 2; ++i) { c3[i].p1 = packw(c3[i].w1, 128, 128); c3[i].p2 = packw(c3[i].w2, 128, 128); }
    d4.p = packw(d4.w, 128, 256);
    for (int i = 0; i < 2; ++i) { c4[i].p1 = packw(c4[i].w1, 256, 256); c4[i].p2 = packw(c4[i].w2, 256, 256); }
    d5.p = packw(d5.w, 256, 256);
    for (int i = 0; i < 2; ++i) { c5[i].p1 = packw(c5[i].w1, 256, 256); c5[i].p2 = packw(c5[i].w2, 256, 256); }
    d6.p = packw(d6.w, 256, 256);
    for (int i = 0; i < 2; ++i) { c6[i].p1 = packw(c6[i].w1, 256, 256); c6[i].p2 = packw(c6[i].w2, 256, 256); }
    unsigned short* pOut = packw(wout_w, 256, 256);
    unsigned short* pSh  = packw(sh_w, 256, 256);

    const int H1 = 736, H2 = 368, H3 = 184, H4 = 92, H5 = 46, H6 = 23;
    const size_t HW1 = (size_t)H1 * H1, HW2 = (size_t)H2 * H2, HW3 = (size_t)H3 * H3;
    const size_t HW4 = (size_t)H4 * H4, HW5 = (size_t)H5 * H5, HW6 = (size_t)H6 * H6;

    bf16_t* xin  = (bf16_t*)alloc(32 * HW1 * 2);    // bf16 NHWC ingest
    bf16_t* big1 = (bf16_t*)alloc(32 * HW1 * 2);
    bf16_t* big2 = (bf16_t*)alloc(32 * HW1 * 2);
    bf16_t* big3 = (bf16_t*)alloc(32 * HW1 * 2);
    bf16_t* mbuf = (bf16_t*)alloc(32 * HW1 * 2);    // x*kept view for dyn convs
    bf16_t* s4a = (bf16_t*)alloc(256 * HW4 * 2);
    bf16_t* s4b = (bf16_t*)alloc(256 * HW4 * 2);
    bf16_t* s4c = (bf16_t*)alloc(256 * HW4 * 2);
    bf16_t* s5a = (bf16_t*)alloc(256 * HW5 * 2);
    bf16_t* s5b = (bf16_t*)alloc(256 * HW5 * 2);
    bf16_t* s5c = (bf16_t*)alloc(256 * HW5 * 2);
    bf16_t* s6a = (bf16_t*)alloc(256 * HW6 * 2);
    bf16_t* s6b = (bf16_t*)alloc(256 * HW6 * 2);
    bf16_t* s6c = (bf16_t*)alloc(256 * HW6 * 2);
    bf16_t* xb  = (bf16_t*)alloc(256 * HW4 * 2);
    float* m2 = (float*)alloc(HW2 * 4);
    float* m3 = (float*)alloc(HW3 * 4);
    float* m4 = (float*)alloc(HW4 * 4);
    float* m5 = (float*)alloc(HW5 * 4);
    float* m6 = (float*)alloc(HW6 * 4);
    float* mb = (float*)alloc(HW4 * 4);
    float* mo = (float*)alloc(HW4 * 4);
    float* score  = (float*)alloc(HW1 * 4);
    float* kept   = (float*)alloc(HW1 * 4);
    float* unkept = (float*)alloc(HW1 * 4);
    int*   hist = (int*)alloc(HISTBINS * 4);
    float* cnt  = (float*)alloc(4);
    float* thr  = (float*)alloc(4);

    auto conv = [&](const bf16_t* inC, const bf16_t* inO, unsigned short* wpk,
                    const float* bias, BNP bn, float eps,
                    const bf16_t* res, const float* mOut,
                    bf16_t* outBF, float* outF32,
                    int Cin, int Cout, int Hin, int Hout, int stride) {
        int Mtiles = (Hout * Hout + 15) / 16;
        dim3 g((Mtiles + 3) / 4, Cout / 32);          // 2 N-tiles per wave
        size_t shm = (size_t)6 * (Cin >> 5) * 1024;   // 3 taps x KT x 2 nts
        switch (Cin >> 5) {
        case 1: conv3x3_wmma<1><<<g, 128, shm, stream>>>(inC, inO, wpk, bias,
                    bn.g, bn.b, bn.m, bn.v, eps, res, mOut, outBF, outF32,
                    Cout, Hin, Hin, Hout, Hout, stride); break;
        case 2: conv3x3_wmma<2><<<g, 128, shm, stream>>>(inC, inO, wpk, bias,
                    bn.g, bn.b, bn.m, bn.v, eps, res, mOut, outBF, outF32,
                    Cout, Hin, Hin, Hout, Hout, stride); break;
        case 4: conv3x3_wmma<4><<<g, 128, shm, stream>>>(inC, inO, wpk, bias,
                    bn.g, bn.b, bn.m, bn.v, eps, res, mOut, outBF, outF32,
                    Cout, Hin, Hin, Hout, Hout, stride); break;
        default: conv3x3_wmma<8><<<g, 128, shm, stream>>>(inC, inO, wpk, bias,
                    bn.g, bn.b, bn.m, bn.v, eps, res, mOut, outBF, outF32,
                    Cout, Hin, Hin, Hout, Hout, stride); break;
        }
    };

    auto runbb = [&](BBP& p, const bf16_t* xin_, bf16_t* t, bf16_t* outb,
                     const float* mask, int C, int H) {
        conv(xin_, xin_, p.p1, p.b1, p.bn1, 1e-3f, nullptr, mask, t, nullptr, C, C, H, H, 1);
        conv(t, t,       p.p2, p.b2, p.bn2, 1e-3f, xin_,    mask, outb, nullptr, C, C, H, H, 1);
    };

    auto dyn = [&](DWP& d, const bf16_t* xin_, const float* mIn, float* mNew,
                   bf16_t* outb, int Cin, int Cout, int Hin, int Hout) {
        size_t HWi = (size_t)Hin * Hin, HWo = (size_t)Hout * Hout;
        hipMemsetAsync(hist, 0, HISTBINS * 4, stream);
        hipMemsetAsync(cnt, 0, 4, stream);
        score_k<<<dim3((HWi + 255) / 256), dim3(256), 0, stream>>>(
            xin_, mIn, d.wat, d.bat, score, hist, cnt, Cin, (int)HWi);
        thresh_k<<<dim3(1), dim3(32), 0, stream>>>(hist, cnt, 0.5f, thr);
        kept_k<<<dim3((HWi + 255) / 256), dim3(256), 0, stream>>>(
            mIn, score, thr, kept, unkept, (int)HWi);
        dynmask_k<<<dim3((HWo + 255) / 256), dim3(256), 0, stream>>>(
            kept, unkept, mNew, Hin, Hout);
        long tot = (long)HWi * Cin;
        maskmul_k<<<dim3((unsigned)((tot + 255) / 256)), dim3(256), 0, stream>>>(
            xin_, kept, mbuf, Cin, tot);
        conv(xin_, mbuf, d.p, nullptr, d.bn, 1e-3f, nullptr, mNew, outb, nullptr,
             Cin, Cout, Hin, Hout, 2);
    };

    auto sp = [&](DWP& d, const bf16_t* xin_, const float* mIn, float* mNew,
                  bf16_t* outb, int C, int Hin, int Hout) {
        dilate_k<<<dim3((Hout * Hout + 255) / 256), dim3(256), 0, stream>>>(
            mIn, mNew, Hin, Hout, 2);
        conv(xin_, xin_, d.p, nullptr, d.bn, 1e-3f, nullptr, mNew, outb, nullptr,
             C, C, Hin, Hout, 2);
    };

    // ---- ingest: fp32 NCHW -> bf16 NHWC ----------------------------------
    {
        long tot = (long)32 * HW1;
        nchw2nhwc_bf16_k<<<dim3((unsigned)((tot + 255) / 256)), dim3(256), 0, stream>>>(
            x0f, xin, 32, (long)HW1);
    }

    // ---- stage 1 (stride 1, C=32) ----------------------------------------
    runbb(c1[0], xin,  big1, big2, mask1, 32, H1);
    runbb(c1[1], big2, big1, big3, mask1, 32, H1);          // x1 = big3

    // ---- stage 2 (dyn down 32->64) ---------------------------------------
    dyn(d2, big3, mask1, m2, big1, 32, 64, H1, H2);
    runbb(c2[0], big1, big2, big3, m2, 64, H2);
    runbb(c2[1], big3, big2, big1, m2, 64, H2);             // x2 = big1

    // ---- stage 3 (dyn down 64->128) --------------------------------------
    dyn(d3, big1, m2, m3, big2, 64, 128, H2, H3);
    runbb(c3[0], big2, big3, big1, m3, 128, H3);
    runbb(c3[1], big1, big3, big2, m3, 128, H3);            // x3 = big2

    // ---- stage 4 (dyn down 128->256) -------------------------------------
    dyn(d4, big2, m3, m4, s4a, 128, 256, H3, H4);
    runbb(c4[0], s4a, s4b, s4c, m4, 256, H4);
    runbb(c4[1], s4c, s4b, s4a, m4, 256, H4);               // x4 = s4a

    // ---- stage 5 (spconv down) -------------------------------------------
    sp(d5, s4a, m4, m5, s5a, 256, H4, H5);
    runbb(c5[0], s5a, s5b, s5c, m5, 256, H5);
    runbb(c5[1], s5c, s5b, s5a, m5, 256, H5);               // x5 = s5a

    // ---- stage 6 (spconv down) -------------------------------------------
    sp(d6, s5a, m5, m6, s6a, 256, H5, H6);
    runbb(c6[0], s6a, s6b, s6c, m6, 256, H6);
    runbb(c6[1], s6c, s6b, s6a, m6, 256, H6);               // x6 = s6a

    // ---- BEV combine + output convs --------------------------------------
    bev_add_k<<<dim3((unsigned)((256 * HW4 + 255) / 256)), dim3(256), 0, stream>>>(
        s4a, s5a, s6a, xb, 256, H4);
    bev_mask_k<<<dim3((unsigned)((HW4 + 255) / 256)), dim3(256), 0, stream>>>(
        m4, m5, m6, mb, H4);
    dilate_k<<<dim3((unsigned)((HW4 + 255) / 256)), dim3(256), 0, stream>>>(
        mb, mo, H4, H4, 1);
    conv(xb, xb,   pOut, nullptr, wout_bn, 1e-3f, nullptr, mo, s4b, nullptr,
         256, 256, H4, H4, 1);
    conv(s4b, s4b, pSh,  sh_b,    sh_bn,   1e-5f, nullptr, mo, nullptr, (float*)d_out,
         256, 256, H4, H4, 1);
}
